// Decoder_without_dropout_75591424410108
// MI455X (gfx1250) — matrix operand order repaired and checked by reading, not compiled, not run on hardware
//
#include <hip/hip_runtime.h>
#include <hip/hip_bf16.h>
#include <cstdint>

typedef __bf16 bf16;
typedef __bf16 v16bf __attribute__((ext_vector_type(16)));
typedef __bf16 v8bf  __attribute__((ext_vector_type(8)));
typedef float  v8f   __attribute__((ext_vector_type(8)));
typedef int    v4i   __attribute__((ext_vector_type(4)));

constexpr int EMB = 1024, HID = 1024, NL = 2, T = 64, B = 128, S = 128;
constexpr int H3 = 3 * HID;

// ---------------- gfx1250 async global->LDS copy (ASYNCcnt path) ----------------
#if __has_builtin(__builtin_amdgcn_global_load_async_to_lds_b128)
#define GEMM_ASYNC 1
typedef __attribute__((address_space(1))) v4i* gas_v4i;
typedef __attribute__((address_space(3))) v4i* las_v4i;
__device__ __forceinline__ void async_cp16(const bf16* g, bf16* l) {
  __builtin_amdgcn_global_load_async_to_lds_b128(
      (gas_v4i)(uintptr_t)g, (las_v4i)l, 0, 0);
}
#else
#define GEMM_ASYNC 0
#endif

#if GEMM_ASYNC
__device__ __forceinline__ void wait_async_4() {
#if __has_builtin(__builtin_amdgcn_s_wait_asynccnt)
  __builtin_amdgcn_s_wait_asynccnt(4);
#else
  asm volatile("s_wait_asynccnt 0x4" ::: "memory");
#endif
}
__device__ __forceinline__ void wait_async_0() {
#if __has_builtin(__builtin_amdgcn_s_wait_asynccnt)
  __builtin_amdgcn_s_wait_asynccnt(0);
#else
  asm volatile("s_wait_asynccnt 0x0" ::: "memory");
#endif
}
#endif

// ---------------- fp32 -> bf16 ----------------
__global__ void cvt_f32_bf16(const float* __restrict__ src, bf16* __restrict__ dst, int n) {
  int i = blockIdx.x * blockDim.x + threadIdx.x;
  if (i < n) dst[i] = (bf16)src[i];
}

// ---------------- embedding gather -> bf16 ----------------
__global__ void embed_gather(const int* __restrict__ tok, const float* __restrict__ emb,
                             bf16* __restrict__ X) {
  int i = blockIdx.x * blockDim.x + threadIdx.x;   // over T*B*EMB
  int row = i >> 10, e = i & (EMB - 1);
  int t = tok[row];
  X[i] = (bf16)emb[(size_t)t * EMB + e];
}

// ---------------- bf16 WMMA GEMM:  out(M,N) = A(M,K) @ W(N,K)^T (+bias, +act) ----
// A may be split along K:  k < Ksplit -> A (stride ldA), k >= Ksplit -> A2 (stride ldA2).
// Ksplit is a multiple of 32, so the split select is uniform per K-step.
// Workgroup tile 128x128, K-step 32, double-buffered LDS staged with async copies;
// 8 waves each compute a 32x64 tile via 8 WMMA accumulators.
union Frag { v16bf v; struct Halves { v8bf lo, hi; } p; };

template <int ACT, bool OUTBF>
__global__ void __launch_bounds__(256)
gemm_bf16(const bf16* __restrict__ A, int ldA,
          const bf16* __restrict__ A2, int ldA2, int Ksplit,
          const bf16* __restrict__ W, const float* __restrict__ bias,
          void* __restrict__ outp, int N, int K) {
  constexpr int LDT = 40;                 // padded row (80B) to spread LDS banks
  __shared__ __align__(16) bf16 As[2][128 * LDT];
  __shared__ __align__(16) bf16 Bs[2][128 * LDT];

  const int m0 = blockIdx.y * 128, n0 = blockIdx.x * 128;
  const int tid = threadIdx.x;
  const int wave = tid >> 5, lane = tid & 31;
  const int mw = (wave >> 1) * 32, nw = (wave & 1) * 64;
  const int r16 = lane & 15, hf = lane >> 4;

  // staging geometry: each thread copies rows r0 and r0+64 at column cc (16B chunks)
  const int r0 = tid >> 2;
  const int cc = (tid & 3) * 8;
  const bf16* A2s = A2 ? A2 : A;
  const unsigned offA  = (unsigned)(m0 + r0) * (unsigned)ldA  + (unsigned)cc;
  const unsigned offA2 = (unsigned)(m0 + r0) * (unsigned)ldA2 + (unsigned)cc;
  const unsigned offW  = (unsigned)(n0 + r0) * (unsigned)K    + (unsigned)cc;
  const unsigned strA  = 64u * (unsigned)ldA;
  const unsigned strA2 = 64u * (unsigned)ldA2;
  const unsigned strW  = 64u * (unsigned)K;

  auto stage = [&](int buf, int k0) {
    const bf16 *g0, *g1;
    if (k0 < Ksplit) {                       // uniform branch (Ksplit % 32 == 0, cc < 32)
      g0 = A + (offA + (unsigned)k0);
      g1 = g0 + strA;
    } else {
      g0 = A2s + (offA2 + (unsigned)(k0 - Ksplit));
      g1 = g0 + strA2;
    }
    const bf16* g2 = W + (offW + (unsigned)k0);
    const bf16* g3 = g2 + strW;
    bf16* l0 = &As[buf][r0 * LDT + cc];
    bf16* l1 = &As[buf][(r0 + 64) * LDT + cc];
    bf16* l2 = &Bs[buf][r0 * LDT + cc];
    bf16* l3 = &Bs[buf][(r0 + 64) * LDT + cc];
#if GEMM_ASYNC
    async_cp16(g0, l0);
    async_cp16(g1, l1);
    async_cp16(g2, l2);
    async_cp16(g3, l3);
#else
    uint4 t0 = *(const uint4*)g0;
    uint4 t1 = *(const uint4*)g1;
    uint4 t2 = *(const uint4*)g2;
    uint4 t3 = *(const uint4*)g3;
    *(uint4*)l0 = t0;
    *(uint4*)l1 = t1;
    *(uint4*)l2 = t2;
    *(uint4*)l3 = t3;
#endif
  };

  const v8f vzero = {0.f, 0.f, 0.f, 0.f, 0.f, 0.f, 0.f, 0.f};
  v8f acc[2][4];
  for (int mi = 0; mi < 2; ++mi)
    for (int ni = 0; ni < 4; ++ni) acc[mi][ni] = vzero;

  const int nIter = K / 32;
  stage(0, 0);
  for (int it = 0; it < nIter; ++it) {
    const int buf = it & 1;
    const bool more = (it + 1 < nIter);
    if (more) stage(buf ^ 1, (it + 1) * 32);   // prefetch next tile into other buffer
#if GEMM_ASYNC
    if (more) wait_async_4();                  // current tile's 4 copies done
    else      wait_async_0();
#endif
    __syncthreads();

    Frag af[2], bfg[4];
    for (int mi = 0; mi < 2; ++mi) {
      int row = mw + mi * 16 + r16;
      af[mi].p.lo = *(const v8bf*)&As[buf][row * LDT + hf * 8];       // K = 8h..8h+7
      af[mi].p.hi = *(const v8bf*)&As[buf][row * LDT + 16 + hf * 8];  // K = 16+8h..
    }
    for (int ni = 0; ni < 4; ++ni) {
      int nn = nw + ni * 16 + r16;
      bfg[ni].p.lo = *(const v8bf*)&Bs[buf][nn * LDT + hf * 8];       // K = 8h..8h+7
      bfg[ni].p.hi = *(const v8bf*)&Bs[buf][nn * LDT + 16 + hf * 8];  // K = 16+8h..
    }
    for (int mi = 0; mi < 2; ++mi)
      for (int ni = 0; ni < 4; ++ni)
        acc[mi][ni] = __builtin_amdgcn_wmma_f32_16x16x32_bf16(
            false, af[mi].v, false, bfg[ni].v, (short)0, acc[mi][ni], false, false);
    __syncthreads();   // all waves done reading this buffer before it is restaged
  }

  // epilogue: lane -> N, vgpr j + half*8 -> M
  for (int ni = 0; ni < 4; ++ni) {
    int n_g = n0 + nw + ni * 16 + r16;
    float bv = bias ? bias[n_g] : 0.f;
    for (int mi = 0; mi < 2; ++mi) {
      int mbase = m0 + mw + mi * 16 + hf * 8;
      for (int j = 0; j < 8; ++j) {
        float v = acc[mi][ni][j] + bv;
        if (ACT == 1) v = tanhf(v);
        size_t o = (size_t)(mbase + j) * N + n_g;
        if (OUTBF) ((bf16*)outp)[o] = (bf16)v;
        else       ((float*)outp)[o] = v;
      }
    }
  }
}

// ---------------- GRU gate combine ----------------
__device__ __forceinline__ float sigm_(float x) { return 1.0f / (1.0f + expf(-x)); }

template <bool GIBF>
__global__ void gru_combine(const void* __restrict__ gi_, const float* __restrict__ gh,
                            float* __restrict__ hstate, bf16* __restrict__ hbf) {
  int i = blockIdx.x * blockDim.x + threadIdx.x;   // over B*HID
  int b = i >> 10, j = i & (HID - 1);
  size_t base = (size_t)b * H3;
  float gr, gz, gn;
  if (GIBF) {
    const bf16* gi = (const bf16*)gi_;
    gr = (float)gi[base + j]; gz = (float)gi[base + HID + j]; gn = (float)gi[base + 2 * HID + j];
  } else {
    const float* gi = (const float*)gi_;
    gr = gi[base + j]; gz = gi[base + HID + j]; gn = gi[base + 2 * HID + j];
  }
  float hr = gh[base + j], hz = gh[base + HID + j], hn = gh[base + 2 * HID + j];
  float hp = hstate[i];
  float r = sigm_(gr + hr), z = sigm_(gz + hz);
  float n = tanhf(gn + r * hn);
  float h = (1.f - z) * n + z * hp;
  hstate[i] = h;
  hbf[i] = (bf16)h;
}

// ---------------- attention (tiny; VALU) ----------------
__global__ void attn_scores(const float* __restrict__ Hm, const float* __restrict__ q1,
                            float* __restrict__ sc) {
  int wid = (blockIdx.x * blockDim.x + threadIdx.x) >> 5;
  int lane = threadIdx.x & 31;
  int b = wid >> 7, s = wid & (S - 1);
  const float* hrow = Hm + ((size_t)s * B + b) * HID;   // H is (S,B,HID)
  const float* q = q1 + (size_t)b * HID;
  float acc = 0.f;
  for (int h = lane; h < HID; h += 32) acc += hrow[h] * q[h];
  for (int off = 16; off; off >>= 1) acc += __shfl_xor(acc, off, 32);
  if (lane == 0) sc[b * S + s] = acc;
}

__global__ void attn_softmax(float* __restrict__ sc) {
  __shared__ float red[S];
  int b = blockIdx.x, s = threadIdx.x;
  float v = sc[b * S + s];
  red[s] = v; __syncthreads();
  for (int off = S / 2; off; off >>= 1) { if (s < off) red[s] = fmaxf(red[s], red[s + off]); __syncthreads(); }
  float m = red[0]; __syncthreads();
  float e = expf(v - m);
  red[s] = e; __syncthreads();
  for (int off = S / 2; off; off >>= 1) { if (s < off) red[s] += red[s + off]; __syncthreads(); }
  sc[b * S + s] = e / red[0];
}

__global__ void attn_context(const float* __restrict__ sc, const float* __restrict__ Hm,
                             bf16* __restrict__ cbf) {
  int i = blockIdx.x * blockDim.x + threadIdx.x;   // over B*HID
  int b = i >> 10, h = i & (HID - 1);
  const float* a = sc + b * S;
  float acc = 0.f;
  for (int s = 0; s < S; ++s) acc += a[s] * Hm[((size_t)s * B + b) * HID + h];
  cbf[i] = (bf16)acc;
}

// ---------------- host ----------------
extern "C" void kernel_launch(void* const* d_in, const int* in_sizes, int n_in,
                              void* d_out, int out_size, void* d_ws, size_t ws_size,
                              hipStream_t stream) {
  (void)in_sizes; (void)n_in; (void)out_size; (void)ws_size;
  const int*   input = (const int*)  d_in[0];
  const float* h_in  = (const float*)d_in[1];
  const float* Hm    = (const float*)d_in[2];
  const float* emb   = (const float*)d_in[3];
  const float* w_ih  = (const float*)d_in[4];
  const float* w_hh  = (const float*)d_in[5];
  const float* b_ih  = (const float*)d_in[6];
  const float* b_hh  = (const float*)d_in[7];
  const float* L1    = (const float*)d_in[8];
  const float* L2    = (const float*)d_in[9];
  float* out = (float*)d_out;

  char* p = (char*)d_ws;
  auto alloc = [&](size_t bytes) -> char* {
    char* q = p; p += (bytes + 255) & ~(size_t)255; return q;
  };
  bf16*  w_ih_bf = (bf16*) alloc((size_t)NL * H3 * EMB * 2);
  bf16*  w_hh_bf = (bf16*) alloc((size_t)NL * H3 * HID * 2);
  bf16*  L1_bf   = (bf16*) alloc((size_t)HID * HID * 2);
  bf16*  L2_bf   = (bf16*) alloc((size_t)HID * 2 * HID * 2);
  bf16*  Xemb    = (bf16*) alloc((size_t)T * B * EMB * 2);
  bf16*  GI0     = (bf16*) alloc((size_t)T * B * H3 * 2);
  float* hstate  = (float*)alloc((size_t)NL * B * HID * 4);
  bf16*  hbf     = (bf16*) alloc((size_t)NL * B * HID * 2);
  float* gh0     = (float*)alloc((size_t)B * H3 * 4);
  float* gi1     = (float*)alloc((size_t)B * H3 * 4);
  float* gh1     = (float*)alloc((size_t)B * H3 * 4);
  float* q1      = (float*)alloc((size_t)B * HID * 4);
  float* sc      = (float*)alloc((size_t)B * S * 4);
  bf16*  cbf     = (bf16*) alloc((size_t)B * HID * 2);

  bf16* hbf0 = hbf;
  bf16* hbf1 = hbf + (size_t)B * HID;
  float* h0 = hstate;
  float* h1 = hstate + (size_t)B * HID;

  auto cvt = [&](const float* s, bf16* d, size_t n) {
    cvt_f32_bf16<<<dim3((unsigned)((n + 255) / 256)), 256, 0, stream>>>(s, d, (int)n);
  };
  cvt(w_ih, w_ih_bf, (size_t)NL * H3 * EMB);
  cvt(w_hh, w_hh_bf, (size_t)NL * H3 * HID);
  cvt(L1,   L1_bf,   (size_t)HID * HID);
  cvt(L2,   L2_bf,   (size_t)HID * 2 * HID);
  cvt(h_in, hbf,     (size_t)NL * B * HID);
  (void)hipMemcpyAsync(hstate, h_in, (size_t)NL * B * HID * 4,
                       hipMemcpyDeviceToDevice, stream);

  embed_gather<<<dim3(T * B * EMB / 256), 256, 0, stream>>>(input, emb, Xemb);

  // Parallel precompute: GI0[t,b,:] = embed @ w_ih[0]^T + b_ih[0]   (M=8192, N=3072, K=1024)
  gemm_bf16<0, true><<<dim3(H3 / 128, T * B / 128), 256, 0, stream>>>(
      Xemb, EMB, nullptr, 0, EMB, w_ih_bf, b_ih, GI0, H3, EMB);

  for (int t = 0; t < T; ++t) {
    // layer 0: gh0 = h0 @ w_hh[0]^T + b_hh[0]
    gemm_bf16<0, false><<<dim3(H3 / 128, 1), 256, 0, stream>>>(
        hbf0, HID, nullptr, 0, HID, w_hh_bf, b_hh, gh0, H3, HID);
    gru_combine<true><<<dim3(B * HID / 256), 256, 0, stream>>>(
        GI0 + (size_t)t * B * H3, gh0, h0, hbf0);

    // layer 1: gi1 = h0 @ w_ih[1]^T + b_ih[1] ; gh1 = h1 @ w_hh[1]^T + b_hh[1]
    gemm_bf16<0, false><<<dim3(H3 / 128, 1), 256, 0, stream>>>(
        hbf0, HID, nullptr, 0, HID, w_ih_bf + (size_t)H3 * EMB, b_ih + H3, gi1, H3, EMB);
    gemm_bf16<0, false><<<dim3(H3 / 128, 1), 256, 0, stream>>>(
        hbf1, HID, nullptr, 0, HID, w_hh_bf + (size_t)H3 * HID, b_hh + H3, gh1, H3, HID);
    gru_combine<false><<<dim3(B * HID / 256), 256, 0, stream>>>(gi1, gh1, h1, hbf1);

    // attention: q1 = o @ L1^T ; scores = Hc . q1 ; softmax ; c = a . Hc
    gemm_bf16<0, false><<<dim3(HID / 128, 1), 256, 0, stream>>>(
        hbf1, HID, nullptr, 0, HID, L1_bf, nullptr, q1, HID, HID);
    attn_scores<<<dim3(B * S / 8), 256, 0, stream>>>(Hm, q1, sc);
    attn_softmax<<<dim3(B), S, 0, stream>>>(sc);
    attn_context<<<dim3(B * HID / 256), 256, 0, stream>>>(sc, Hm, cbf);

    // out_t = tanh(concat([c, o]) @ L2^T)  — split-K A operand, tanh epilogue
    gemm_bf16<1, false><<<dim3(HID / 128, 1), 256, 0, stream>>>(
        cbf, HID, hbf1, HID, HID, L2_bf, nullptr, out + (size_t)t * B * HID, HID, 2 * HID);
  }

  // h_final (NL, B, HID)
  (void)hipMemcpyAsync(out + (size_t)T * B * HID, hstate, (size_t)NL * B * HID * 4,
                       hipMemcpyDeviceToDevice, stream);
}
